// FNSAttentionHead_22840636081009
// MI455X (gfx1250) — compile-verified, matrix operand order checked by reading, not run on hardware
//
#include <hip/hip_runtime.h>
#include <hip/hip_bf16.h>

typedef __attribute__((ext_vector_type(16))) _Float16 v16h;
typedef __attribute__((ext_vector_type(8)))  _Float16 v8h;
typedef __attribute__((ext_vector_type(8)))  float    v8f;

#define B_   8
#define S_   2048
#define H_   1024
#define D_   64
#define NROW (B_ * S_)   // 16384

// ---------------------------------------------------------------------------
// Kernel 1: q,k,v = l2norm(x @ W + b).  4 rows per block, thread = (row, d).
// q,k stored fp16 (WMMA operands), v stored fp32.
// ---------------------------------------------------------------------------
__global__ __launch_bounds__(256)
void qkv_kernel(const float* __restrict__ x,
                const float* __restrict__ Wq, const float* __restrict__ bq,
                const float* __restrict__ Wk, const float* __restrict__ bk,
                const float* __restrict__ Wv, const float* __restrict__ bv,
                _Float16* __restrict__ q16, _Float16* __restrict__ k16,
                float* __restrict__ v32)
{
    __shared__ float xs[4 * H_];
    __shared__ float rq[256], rk[256], rv[256];
    const int tid  = threadIdx.x;
    const size_t row0 = (size_t)blockIdx.x * 4;

    for (int i = tid; i < 4 * H_; i += 256)
        xs[i] = x[row0 * H_ + i];
    __syncthreads();

    const int rl = tid >> 6;       // local row 0..3
    const int d  = tid & 63;       // head dim
    const float* xr = xs + rl * H_;
    float aq = 0.f, ak = 0.f, av = 0.f;
    for (int h = 0; h < H_; ++h) {
        const float xv = xr[h];
        aq = fmaf(xv, Wq[h * D_ + d], aq);
        ak = fmaf(xv, Wk[h * D_ + d], ak);
        av = fmaf(xv, Wv[h * D_ + d], av);
    }
    aq += bq[d]; ak += bk[d]; av += bv[d];

    rq[tid] = aq * aq; rk[tid] = ak * ak; rv[tid] = av * av;
    __syncthreads();
    for (int s = 32; s > 0; s >>= 1) {
        if (d < s) {
            rq[tid] += rq[tid + s];
            rk[tid] += rk[tid + s];
            rv[tid] += rv[tid + s];
        }
        __syncthreads();
    }
    const float iq = 1.f / fmaxf(sqrtf(rq[rl * 64]), 1e-12f);
    const float ik = 1.f / fmaxf(sqrtf(rk[rl * 64]), 1e-12f);
    const float iv = 1.f / fmaxf(sqrtf(rv[rl * 64]), 1e-12f);

    const size_t o = (row0 + rl) * D_ + d;
    q16[o] = (_Float16)(aq * iq);
    k16[o] = (_Float16)(ak * ik);
    v32[o] = av * iv;
}

__global__ void zero_kernel(float* __restrict__ p, int n)
{
    const int i = blockIdx.x * 256 + threadIdx.x;
    if (i < n) p[i] = 0.f;
}

// ---------------------------------------------------------------------------
// Kernel 2: score = (1 + arccos(clip(q·kT)))^(-65.5), written straight into
// the probs output region; column sums accumulated via f32 atomics.
// Block = 8 waves; each wave owns a 16x16 tile (2x v_wmma_f32_16x16x32_f16).
// ---------------------------------------------------------------------------
__global__ __launch_bounds__(256)
void score_kernel(const _Float16* __restrict__ q16,
                  const _Float16* __restrict__ k16,
                  float* __restrict__ score,      // d_out probs region [B][S][S]
                  float* __restrict__ colsum)     // [B][S]
{
    const int tid  = threadIdx.x;
    const int lane = tid & 31;
    const int wave = tid >> 5;
    const int bid  = blockIdx.x;
    const int kb   = bid & 15;          // 16 k-blocks of 128 cols
    const int qt   = (bid >> 4) & 127;  // 128 q tiles of 16 rows
    const int b    = bid >> 11;         // batch
    const int qbase = qt * 16;
    const int kcol0 = kb * 128 + wave * 16;

    const int r  = lane & 15;
    const int hi = lane >> 4;
    const size_t qrowbase = ((size_t)b * S_ + qbase + r) * D_;
    const size_t krowbase = ((size_t)b * S_ + kcol0 + r) * D_;

    v8f acc = {};
    #pragma unroll
    for (int c = 0; c < 2; ++c) {
        const size_t off = (size_t)c * 32 + hi * 8;   // halves
        const v8h alo = *(const v8h*)(q16 + qrowbase + off);
        const v8h ahi = *(const v8h*)(q16 + qrowbase + off + 16);
        const v8h blo = *(const v8h*)(k16 + krowbase + off);
        const v8h bhi = *(const v8h*)(k16 + krowbase + off + 16);
        v16h A, Bm;
        #pragma unroll
        for (int i = 0; i < 8; ++i) {
            A[i] = alo[i];  A[i + 8] = ahi[i];
            Bm[i] = blo[i]; Bm[i + 8] = bhi[i];
        }
        acc = __builtin_amdgcn_wmma_f32_16x16x32_f16(
                  false, A, false, Bm, (short)0, acc, false, false);
    }

    const int   col = kcol0 + r;
    const float PI  = 3.14159265358979323846f;
    float colpart = 0.f;
    #pragma unroll
    for (int j = 0; j < 8; ++j) {
        const int m = j + (hi << 3);                 // C layout: M = j (+8 hi half)
        float cosv = acc[j];
        cosv = fminf(fmaxf(cosv, -1.f + 1e-7f), 1.f - 1e-7f);
        const float ax = fabsf(cosv);
        // Abramowitz-Stegun acos approximation on [0,1]
        const float p = fmaf(fmaf(fmaf(-0.0187293f, ax, 0.0742610f), ax,
                                  -0.2121144f), ax, 1.5707288f);
        const float t = sqrtf(1.f - ax) * p;
        const float g = (cosv >= 0.f) ? t : (PI - t);
        // (1+g)^(-HEAD-ALPHA) = exp2(-65.5 * log2(1+g))
        // v_log_f32 is log2, v_exp_f32 is exp2 on AMDGCN
        const float s = __builtin_amdgcn_exp2f(
                            -65.5f * __builtin_amdgcn_logf(1.f + g));
        colpart += s;
        score[((size_t)b * S_ + qbase + m) * S_ + col] = s;
    }
    // combine the two half-tile row-groups (lanes L and L+16 share col)
    const float other = __shfl_down(colpart, 16);
    if (lane < 16)
        atomicAdd(&colsum[b * S_ + col], colpart + other);
}

// Kernel 3: c = colsum^(-0.5)
__global__ void csum_kernel(const float* __restrict__ colsum,
                            float* __restrict__ c, int n)
{
    const int i = blockIdx.x * 256 + threadIdx.x;
    if (i < n) c[i] = rsqrtf(fmaxf(colsum[i], 1e-30f));
}

// ---------------------------------------------------------------------------
// Kernel 4: per row — denom = sum_k score*c ; probs = score*c/denom (in place);
// out = probs @ v fused.  One block per (b,q) row.
// ---------------------------------------------------------------------------
__global__ __launch_bounds__(256)
void probs_out_kernel(float* __restrict__ probs,      // [B][S][S], holds score
                      const float* __restrict__ cbuf, // [B][S]
                      const float* __restrict__ v32,  // [B][S][D]
                      float* __restrict__ outb)       // [B][S][D]
{
    const int tid = threadIdx.x;
    const int b   = blockIdx.x >> 11;
    float* srow = probs + (size_t)blockIdx.x * S_;
    const float* crow = cbuf + b * S_;

    __shared__ float red[256];
    float part = 0.f;
    for (int k = tid; k < S_; k += 256) part += srow[k] * crow[k];
    red[tid] = part;
    __syncthreads();
    for (int s = 128; s > 0; s >>= 1) {
        if (tid < s) red[tid] += red[tid + s];
        __syncthreads();
    }
    const float inv = 1.f / fmaxf(red[0], 1e-12f);

    // in-place rescale to probs (one thread per element; no cross-thread race)
    for (int k = tid; k < S_; k += 256) srow[k] = srow[k] * crow[k] * inv;
    __syncthreads();

    // fused probs @ v : thread = (kidx 0..31, dgrp 0..7)
    const int dgrp = tid & 7;
    const int kidx = tid >> 3;
    float o[8] = {0.f, 0.f, 0.f, 0.f, 0.f, 0.f, 0.f, 0.f};
    const float* vb = v32 + (size_t)b * S_ * D_ + dgrp * 8;
    for (int it = 0; it < S_ / 32; ++it) {
        const int k = kidx + it * 32;
        const float p = srow[k];
        const float* vr = vb + (size_t)k * D_;
        #pragma unroll
        for (int j = 0; j < 8; ++j) o[j] = fmaf(p, vr[j], o[j]);
    }
    __shared__ float lred[256 * 8];
    #pragma unroll
    for (int j = 0; j < 8; ++j) lred[tid * 8 + j] = o[j];
    __syncthreads();
    if (tid < 64) {
        const int dg = tid >> 3, j = tid & 7;
        float s = 0.f;
        for (int kk = 0; kk < 32; ++kk) s += lred[(kk * 8 + dg) * 8 + j];
        outb[(size_t)blockIdx.x * D_ + tid] = s;
    }
}

// ---------------------------------------------------------------------------
extern "C" void kernel_launch(void* const* d_in, const int* in_sizes, int n_in,
                              void* d_out, int out_size, void* d_ws, size_t ws_size,
                              hipStream_t stream)
{
    const float* x  = (const float*)d_in[0];
    const float* Wq = (const float*)d_in[1];
    const float* bq = (const float*)d_in[2];
    const float* Wk = (const float*)d_in[3];
    const float* bk = (const float*)d_in[4];
    const float* Wv = (const float*)d_in[5];
    const float* bv = (const float*)d_in[6];

    char* ws = (char*)d_ws;
    _Float16* q16 = (_Float16*)ws;                         // 2 MB
    _Float16* k16 = (_Float16*)(ws + (2u << 20));          // 2 MB
    float*    v32 = (float*)(ws + (4u << 20));             // 4 MB
    float* colsum = (float*)(ws + (8u << 20));             // 64 KB
    float* cbuf   = (float*)(ws + (8u << 20) + (64u << 10));

    float* outb  = (float*)d_out;                          // [B][S][D]
    float* probs = outb + (size_t)NROW * D_;               // [B][S][S]

    qkv_kernel<<<NROW / 4, 256, 0, stream>>>(x, Wq, bq, Wk, bk, Wv, bv,
                                             q16, k16, v32);
    zero_kernel<<<(NROW + 255) / 256, 256, 0, stream>>>(colsum, NROW);
    score_kernel<<<B_ * 128 * 16, 256, 0, stream>>>(q16, k16, probs, colsum);
    csum_kernel<<<(NROW + 255) / 256, 256, 0, stream>>>(colsum, cbuf, NROW);
    probs_out_kernel<<<NROW, 256, 0, stream>>>(probs, cbuf, v32, outb);
}
